// CausalLocalSDPABlock_23407571763663
// MI455X (gfx1250) — matrix operand order repaired and checked by reading, not compiled, not run on hardware
//
#include <hip/hip_runtime.h>
#include <hip/hip_bf16.h>
#include <math.h>
#include <stdint.h>

// ---------------------------------------------------------------------------
// Types
// ---------------------------------------------------------------------------
typedef __attribute__((ext_vector_type(16))) __bf16 v16bf;
typedef __attribute__((ext_vector_type(8)))  __bf16 bf16x8;
typedef __attribute__((ext_vector_type(8)))  float  v8f;
typedef __attribute__((ext_vector_type(4)))  unsigned int v4u;
typedef __attribute__((ext_vector_type(8)))  int v8i_t;
typedef __attribute__((ext_vector_type(4)))  int v4i_t;

#define D_MODEL  1024
#define N_HEADS  16
#define HEAD_DIM 64
#define D_FF     4096
#define SEQ_L    2048
#define BATCH    2
#define WINDOW   128
#define ROWS     (BATCH * SEQ_L)   // 4096 token rows

#if defined(__gfx1250__) && __has_builtin(__builtin_amdgcn_tensor_load_to_lds)
#define HAVE_TDM 1
#else
#define HAVE_TDM 0
#endif

__device__ __forceinline__ __bf16 f2bf(float f) { return (__bf16)f; }

// Build a 16-element bf16 WMMA fragment from two contiguous 8-element runs.
__device__ __forceinline__ v16bf make_frag(const __bf16* lo, const __bf16* hi) {
    v16bf f;
    bf16x8 a = *(const bf16x8*)lo;
    bf16x8 b = *(const bf16x8*)hi;
#pragma unroll
    for (int i = 0; i < 8; ++i) { f[i] = a[i]; f[i + 8] = b[i]; }
    return f;
}

// ---------------------------------------------------------------------------
// Tensor Data Mover: DMA a 2-D bf16 tile (tile_w x tile_h) from global memory
// (row stride = row_stride elements) into contiguous LDS at lds_off.
// Reads past tensor_w/tensor_h are zero-filled by hardware (OOB rule).
// D# packing per CDNA5 ISA 8.3/8.4.
// ---------------------------------------------------------------------------
__device__ __forceinline__ void tdm_load_2d(unsigned lds_off, const void* gaddr,
                                            unsigned tile_w, unsigned tile_h,
                                            unsigned tensor_w, unsigned tensor_h,
                                            unsigned long long row_stride) {
#if HAVE_TDM
    unsigned long long ga = (unsigned long long)(uintptr_t)gaddr;
    v4u g0;
    g0[0] = 1u;                                            // count = 1 (valid D#)
    g0[1] = lds_off;                                       // lds_addr
    g0[2] = (unsigned)ga;                                  // global_addr[31:0]
    g0[3] = (unsigned)((ga >> 32) & 0x01FFFFFFu) | (2u << 30); // addr[56:32] | type=2
    v8i_t g1;
    g1[0] = (int)(1u << 16);                               // data_size = 1 (2 bytes)
    g1[1] = (int)((tensor_w & 0xFFFFu) << 16);             // tensor_dim0[15:0]
    g1[2] = (int)((tensor_w >> 16) | ((tensor_h & 0xFFFFu) << 16));
    g1[3] = (int)((tensor_h >> 16) | ((tile_w & 0xFFFFu) << 16));
    g1[4] = (int)(tile_h & 0xFFFFu);                       // tile_dim1 (tile_dim2=0)
    g1[5] = (int)(unsigned)(row_stride & 0xFFFFFFFFull);   // tensor_dim0_stride lo
    g1[6] = (int)(unsigned)((row_stride >> 32) & 0xFFFFull);
    g1[7] = 0;
    v4i_t gz = {0, 0, 0, 0};
#if __clang_major__ >= 23
    v8i_t gz8 = {0, 0, 0, 0, 0, 0, 0, 0};
    __builtin_amdgcn_tensor_load_to_lds(g0, g1, gz, gz, gz8, 0);
#else
    __builtin_amdgcn_tensor_load_to_lds(g0, g1, gz, gz, 0);
#endif
#else
    (void)lds_off; (void)gaddr; (void)tile_w; (void)tile_h;
    (void)tensor_w; (void)tensor_h; (void)row_stride;
#endif
}

template<int N>
__device__ __forceinline__ void wait_tensor() {
#if HAVE_TDM
#if __has_builtin(__builtin_amdgcn_s_wait_tensorcnt)
    __builtin_amdgcn_s_wait_tensorcnt(N);
#else
    asm volatile("s_wait_tensorcnt %0" :: "i"(N) : "memory");
#endif
#endif
}

// ---------------------------------------------------------------------------
// Weight transpose + f32 -> bf16 convert:   Wt[n][k] = (bf16) W[k][n]
// ---------------------------------------------------------------------------
__global__ void transpose_bf16_kernel(const float* __restrict__ W,
                                      __bf16* __restrict__ Wt, int K, int N) {
    __shared__ float tile[32][33];
    const int n0 = blockIdx.x * 32;
    const int k0 = blockIdx.y * 32;
    for (int i = threadIdx.y; i < 32; i += 8)
        tile[i][threadIdx.x] = W[(size_t)(k0 + i) * N + n0 + threadIdx.x];
    __syncthreads();
    for (int i = threadIdx.y; i < 32; i += 8)
        Wt[(size_t)(n0 + i) * K + k0 + threadIdx.x] = f2bf(tile[threadIdx.x][i]);
}

// ---------------------------------------------------------------------------
// LayerNorm (f32 in) -> bf16 out.  One block (256 thr) per 1024-wide row.
// ---------------------------------------------------------------------------
__global__ void ln_bf16_kernel(const float* __restrict__ x,
                               const float* __restrict__ w,
                               const float* __restrict__ b,
                               __bf16* __restrict__ out) {
    const int row = blockIdx.x, t = threadIdx.x;
    const float* xr = x + (size_t)row * D_MODEL;
    float4 v = ((const float4*)xr)[t];
    float s  = v.x + v.y + v.z + v.w;
    float sq = v.x * v.x + v.y * v.y + v.z * v.z + v.w * v.w;
    __shared__ float rs[256], rq[256];
    rs[t] = s; rq[t] = sq;
    __syncthreads();
    for (int off = 128; off > 0; off >>= 1) {
        if (t < off) { rs[t] += rs[t + off]; rq[t] += rq[t + off]; }
        __syncthreads();
    }
    const float mu   = rs[0] * (1.0f / D_MODEL);
    const float var  = rq[0] * (1.0f / D_MODEL) - mu * mu;
    const float rstd = rsqrtf(var + 1e-5f);
    float4 wv = ((const float4*)w)[t];
    float4 bv = ((const float4*)b)[t];
    __bf16* o = out + (size_t)row * D_MODEL + t * 4;
    o[0] = f2bf((v.x - mu) * rstd * wv.x + bv.x);
    o[1] = f2bf((v.y - mu) * rstd * wv.y + bv.y);
    o[2] = f2bf((v.z - mu) * rstd * wv.z + bv.z);
    o[3] = f2bf((v.w - mu) * rstd * wv.w + bv.w);
}

// ---------------------------------------------------------------------------
// bf16 WMMA GEMM:  C[M][N] = A[M][K] * Wt[N][K]^T + bias  (+ epilogue)
//   EPI 0: store bf16            (QKV projections)
//   EPI 1: += resid, store f32   (O-proj, FFN2)
//   EPI 2: gelu, store bf16      (FFN1)
// Block tile 64x64, 128 threads = 4 waves, each wave -> 16 rows x 64 cols.
// K is consumed in 64-wide stages, double-buffered through LDS by the Tensor
// Data Mover: wave 0 keeps two stages in flight (s_wait_tensorcnt 2) so the
// DMA of stage s+2 overlaps the WMMAs of stage s+1.
// ---------------------------------------------------------------------------
template<int EPI>
__global__ void gemm_bf16_kernel(const __bf16* __restrict__ A,
                                 const __bf16* __restrict__ Wt,
                                 const float*  __restrict__ bias,
                                 const float*  __restrict__ resid,
                                 float*        __restrict__ outf,
                                 __bf16*       __restrict__ outb,
                                 int M, int N, int K) {
    __shared__ __bf16 sA[2][64][64];   // 16 KB
    __shared__ __bf16 sB[2][64][64];   // 16 KB
    const int tid  = threadIdx.x;
    const int wave = tid >> 5;
    const int lane = tid & 31;
    const int m0   = blockIdx.y * 64;
    const int n0   = blockIdx.x * 64;

    v8f acc[4] = {};

    const int am = (wave << 4) + (lane & 15);
    const int ko = (lane >> 4) << 3;   // 0 or 8
    const int kg = (lane >> 4) << 4;   // 0 or 16
    const int nstage = K >> 6;

#if HAVE_TDM
    if (wave == 0) {
        tdm_load_2d((unsigned)(uintptr_t)&sA[0][0][0], &A [(size_t)m0 * K],
                    64, 64, 64, 64, (unsigned long long)K);
        tdm_load_2d((unsigned)(uintptr_t)&sB[0][0][0], &Wt[(size_t)n0 * K],
                    64, 64, 64, 64, (unsigned long long)K);
        if (nstage > 1) {
            tdm_load_2d((unsigned)(uintptr_t)&sA[1][0][0], &A [(size_t)m0 * K + 64],
                        64, 64, 64, 64, (unsigned long long)K);
            tdm_load_2d((unsigned)(uintptr_t)&sB[1][0][0], &Wt[(size_t)n0 * K + 64],
                        64, 64, 64, 64, (unsigned long long)K);
            wait_tensor<2>();          // stage 0 landed, stage 1 still in flight
        } else {
            wait_tensor<0>();
        }
    }
    __syncthreads();
#endif

    for (int s = 0; s < nstage; ++s) {
        const int cur = s & 1;
#if !HAVE_TDM
        {   // synchronous fallback staging
            const int k0 = s << 6;
            const int r  = tid >> 1;
            const int cb = (tid & 1) << 5;
#pragma unroll
            for (int c = 0; c < 4; ++c) {
                *(bf16x8*)&sA[cur][r][cb + c * 8] =
                    *(const bf16x8*)&A [(size_t)(m0 + r) * K + k0 + cb + c * 8];
                *(bf16x8*)&sB[cur][r][cb + c * 8] =
                    *(const bf16x8*)&Wt[(size_t)(n0 + r) * K + k0 + cb + c * 8];
            }
            __syncthreads();
        }
#endif
#pragma unroll
        for (int ch = 0; ch < 2; ++ch) {          // 64-wide stage = 2 x k32
            v16bf afrag = make_frag(&sA[cur][am][ch * 32 + ko],
                                    &sA[cur][am][ch * 32 + ko + 16]);
#pragma unroll
            for (int nt = 0; nt < 4; ++nt) {
                const int bn = (nt << 4) + (lane & 15);
                v16bf bfrag = make_frag(&sB[cur][bn][ch * 32 + ko],
                                        &sB[cur][bn][ch * 32 + ko + 16]);
                acc[nt] = __builtin_amdgcn_wmma_f32_16x16x32_bf16(
                    false, afrag, false, bfrag, (short)0, acc[nt], false, false);
            }
        }
        __syncthreads();                          // readers done with buf[cur]
#if HAVE_TDM
        if (wave == 0) {
            if (s + 2 < nstage) {
                const int kn = (s + 2) << 6;
                tdm_load_2d((unsigned)(uintptr_t)&sA[cur][0][0],
                            &A [(size_t)m0 * K + kn], 64, 64, 64, 64,
                            (unsigned long long)K);
                tdm_load_2d((unsigned)(uintptr_t)&sB[cur][0][0],
                            &Wt[(size_t)n0 * K + kn], 64, 64, 64, 64,
                            (unsigned long long)K);
                wait_tensor<2>();  // stage s+1 complete, stage s+2 in flight
            } else {
                wait_tensor<0>();  // drain
            }
        }
        if (((s + 2) << 6) < K) {  // warm L2 for stage s+2 (same stage the TDM fetches)
            const int kn = (s + 2) << 6;
            __builtin_prefetch(&A [(size_t)(m0 + (tid >> 1)) * K + kn + ((tid & 1) << 5)], 0, 1);
            __builtin_prefetch(&Wt[(size_t)(n0 + (tid >> 1)) * K + kn + ((tid & 1) << 5)], 0, 1);
        }
        __syncthreads();                          // buf[next] now valid for all
#endif
    }

    // Epilogue.  C layout: element r of v8f -> row = r + 8*(lane>>4), col = lane&15.
#pragma unroll
    for (int nt = 0; nt < 4; ++nt) {
        const int col = n0 + (nt << 4) + (lane & 15);
        const float bv = bias[col];
#pragma unroll
        for (int r = 0; r < 8; ++r) {
            const int row = m0 + (wave << 4) + r + ((lane >> 4) << 3);
            const size_t idx = (size_t)row * N + col;
            float v = acc[nt][r] + bv;
            if (EPI == 1) {
                outf[idx] = v + resid[idx];
            } else if (EPI == 2) {
                float t = 0.7978845608f * (v + 0.044715f * v * v * v);
                outb[idx] = f2bf(0.5f * v * (1.0f + tanhf(t)));
            } else {
                outb[idx] = f2bf(v);
            }
        }
    }
}

// ---------------------------------------------------------------------------
// Sliding-window causal attention, flash-style in LDS.
// One block per (batch, head, 16-query tile); 128 threads = 4 waves.
// Keys needed: [max(0, q0-128), q0+15]  ->  <=144, padded to 160 (10 tiles).
// Q and K tiles staged by TDM (K uses hardware OOB zero-fill for padding);
// V is manually transposed into LDS so P*V B-fragments are contiguous.
// ---------------------------------------------------------------------------
#define KPAD 160

__global__ void attn_kernel(const __bf16* __restrict__ qmat,
                            const __bf16* __restrict__ kmat,
                            const __bf16* __restrict__ vmat,
                            __bf16* __restrict__ ctx) {
    __shared__ __bf16 sQ[16][HEAD_DIM];       //  2 KB
    __shared__ __bf16 sK[KPAD][HEAD_DIM];     // 20 KB  (key-major, row = key)
    __shared__ __bf16 sVt[HEAD_DIM][KPAD];    // 20 KB  (dim-major: V^T)
    __shared__ float  sS[16][KPAD];           // 10 KB
    __shared__ __bf16 sP[16][KPAD];           //  5 KB

    const int qb = blockIdx.x, h = blockIdx.y, bz = blockIdx.z;
    const int tid  = threadIdx.x;
    const int wave = tid >> 5;
    const int lane = tid & 31;

    const int q0  = qb * 16;
    const int jlo = (q0 >= WINDOW) ? (q0 - WINDOW) : 0;
    const int nk  = q0 + 16 - jlo;            // valid keys (<=144)

    const __bf16* qg  = qmat + ((size_t)bz * SEQ_L) * D_MODEL + h * HEAD_DIM;
    const __bf16* kg_ = kmat + ((size_t)bz * SEQ_L) * D_MODEL + h * HEAD_DIM;
    const __bf16* vg_ = vmat + ((size_t)bz * SEQ_L) * D_MODEL + h * HEAD_DIM;

    // --- stage Q (16x64), K (zero-padded to 160x64), V^T (64x160) ---------
#if HAVE_TDM
    if (wave == 0) {
        tdm_load_2d((unsigned)(uintptr_t)&sQ[0][0], &qg[(size_t)q0 * D_MODEL],
                    HEAD_DIM, 16, HEAD_DIM, 16, (unsigned long long)D_MODEL);
        // tensor_dim1 = nk: rows beyond the valid window are zero-filled (OOB)
        tdm_load_2d((unsigned)(uintptr_t)&sK[0][0], &kg_[(size_t)jlo * D_MODEL],
                    HEAD_DIM, KPAD, HEAD_DIM, (unsigned)nk, (unsigned long long)D_MODEL);
    }
#else
    {
        const int qr = tid >> 3, qc = (tid & 7) << 3;
        *(bf16x8*)&sQ[qr][qc] = *(const bf16x8*)&qg[(size_t)(q0 + qr) * D_MODEL + qc];
    }
#endif
    for (int c = tid; c < KPAD * 8; c += 128) {
        const int krow = c >> 3;
        const int ch   = (c & 7) << 3;
        bf16x8 vv = {};
#if !HAVE_TDM
        bf16x8 kv = {};
#endif
        if (krow < nk) {
            const size_t off = (size_t)(jlo + krow) * D_MODEL + ch;
            vv = *(const bf16x8*)&vg_[off];
#if !HAVE_TDM
            kv = *(const bf16x8*)&kg_[off];
#endif
        }
#if !HAVE_TDM
        *(bf16x8*)&sK[krow][ch] = kv;
#endif
#pragma unroll
        for (int i = 0; i < 8; ++i) sVt[ch + i][krow] = vv[i];
    }
#if HAVE_TDM
    if (wave == 0) wait_tensor<0>();
#endif
    __syncthreads();

    // --- scores: S = (Q K^T) * scale, masked ------------------------------
    const float scale = 0.125f;   // 1/sqrt(64)
    for (int kt = wave; kt < KPAD / 16; kt += 4) {
        v8f acc = {};
#pragma unroll
        for (int ch = 0; ch < 2; ++ch) {      // HEAD_DIM = 2 * 32
            const int ko2 = ((lane >> 4) << 3) + ch * 32;
            const int kg2 = ((lane >> 4) << 4) + ch * 32;
            v16bf afrag = make_frag(&sQ[lane & 15][ko2], &sQ[lane & 15][ko2 + 16]);
            v16bf bfrag = make_frag(&sK[kt * 16 + (lane & 15)][ko2],
                                    &sK[kt * 16 + (lane & 15)][ko2 + 16]);
            acc = __builtin_amdgcn_wmma_f32_16x16x32_bf16(
                false, afrag, false, bfrag, (short)0, acc, false, false);
        }
        const int keycol = kt * 16 + (lane & 15);
        const int j = jlo + keycol;
#pragma unroll
        for (int r = 0; r < 8; ++r) {
            const int qrow = r + ((lane >> 4) << 3);
            const int i = q0 + qrow;
            const bool valid = (keycol < nk) && (j <= i) && ((i - j) <= WINDOW);
            sS[qrow][keycol] = valid ? acc[r] * scale : -1e30f;
        }
    }
    __syncthreads();

    // --- softmax over keys (16 rows, 8 threads/row) -----------------------
    {
        const int row = tid >> 3, c0 = tid & 7;
        float m = -1e30f;
        for (int c = c0; c < KPAD; c += 8) m = fmaxf(m, sS[row][c]);
        m = fmaxf(m, __shfl_xor(m, 1, 8));
        m = fmaxf(m, __shfl_xor(m, 2, 8));
        m = fmaxf(m, __shfl_xor(m, 4, 8));
        float s = 0.0f;
        for (int c = c0; c < KPAD; c += 8) s += __expf(sS[row][c] - m);
        s += __shfl_xor(s, 1, 8);
        s += __shfl_xor(s, 2, 8);
        s += __shfl_xor(s, 4, 8);
        const float inv = 1.0f / s;
        for (int c = c0; c < KPAD; c += 8)
            sP[row][c] = f2bf(__expf(sS[row][c] - m) * inv);
    }
    __syncthreads();

    // --- ctx = P V  (each wave owns one 16-wide slice of HEAD_DIM) --------
    {
        v8f acc = {};
        const int dim = (wave << 4) + (lane & 15);
#pragma unroll
        for (int c = 0; c < KPAD / 32; ++c) {
            const int ko2 = ((lane >> 4) << 3) + c * 32;
            const int kg2 = ((lane >> 4) << 4) + c * 32;
            v16bf afrag = make_frag(&sP[lane & 15][ko2], &sP[lane & 15][ko2 + 16]);
            v16bf bfrag = make_frag(&sVt[dim][ko2], &sVt[dim][ko2 + 16]);
            acc = __builtin_amdgcn_wmma_f32_16x16x32_bf16(
                false, afrag, false, bfrag, (short)0, acc, false, false);
        }
        __bf16* cg = ctx + ((size_t)bz * SEQ_L) * D_MODEL + h * HEAD_DIM;
#pragma unroll
        for (int r = 0; r < 8; ++r) {
            const int qrow = r + ((lane >> 4) << 3);
            cg[(size_t)(q0 + qrow) * D_MODEL + dim] = f2bf(acc[r]);
        }
    }
}

// ---------------------------------------------------------------------------
// Orchestration
// ---------------------------------------------------------------------------
extern "C" void kernel_launch(void* const* d_in, const int* in_sizes, int n_in,
                              void* d_out, int out_size, void* d_ws, size_t ws_size,
                              hipStream_t stream) {
    const float* x    = (const float*)d_in[0];
    const float* Wq   = (const float*)d_in[1];
    const float* bq   = (const float*)d_in[2];
    const float* Wk   = (const float*)d_in[3];
    const float* bk   = (const float*)d_in[4];
    const float* Wv   = (const float*)d_in[5];
    const float* bv   = (const float*)d_in[6];
    const float* Wo   = (const float*)d_in[7];
    const float* bo   = (const float*)d_in[8];
    const float* W1   = (const float*)d_in[9];
    const float* b1   = (const float*)d_in[10];
    const float* W2   = (const float*)d_in[11];
    const float* b2   = (const float*)d_in[12];
    const float* ln1w = (const float*)d_in[13];
    const float* ln1b = (const float*)d_in[14];
    const float* ln2w = (const float*)d_in[15];
    const float* ln2b = (const float*)d_in[16];

    // Workspace layout (bf16 elements unless noted); ~120 MB total.
    __bf16* WqT = (__bf16*)d_ws;
    __bf16* WkT = WqT + (1u << 20);
    __bf16* WvT = WkT + (1u << 20);
    __bf16* WoT = WvT + (1u << 20);
    __bf16* W1T = WoT + (1u << 20);    // [D_FF][D]   4M
    __bf16* W2T = W1T + (4u << 20);    // [D][D_FF]   4M
    __bf16* h1  = W2T + (4u << 20);    // [4096][1024]
    __bf16* qb_ = h1  + (4u << 20);
    __bf16* kb_ = qb_ + (4u << 20);
    __bf16* vb_ = kb_ + (4u << 20);
    __bf16* ctb = vb_ + (4u << 20);
    __bf16* h2  = ctb + (4u << 20);
    __bf16* gb_ = h2  + (4u << 20);    // [4096][4096] 16M
    float*  x1  = (float*)(gb_ + (16u << 20));

    const dim3 tb(32, 8);
    transpose_bf16_kernel<<<dim3(D_MODEL/32, D_MODEL/32), tb, 0, stream>>>(Wq, WqT, D_MODEL, D_MODEL);
    transpose_bf16_kernel<<<dim3(D_MODEL/32, D_MODEL/32), tb, 0, stream>>>(Wk, WkT, D_MODEL, D_MODEL);
    transpose_bf16_kernel<<<dim3(D_MODEL/32, D_MODEL/32), tb, 0, stream>>>(Wv, WvT, D_MODEL, D_MODEL);
    transpose_bf16_kernel<<<dim3(D_MODEL/32, D_MODEL/32), tb, 0, stream>>>(Wo, WoT, D_MODEL, D_MODEL);
    transpose_bf16_kernel<<<dim3(D_FF/32,    D_MODEL/32), tb, 0, stream>>>(W1, W1T, D_MODEL, D_FF);
    transpose_bf16_kernel<<<dim3(D_MODEL/32, D_FF/32),    tb, 0, stream>>>(W2, W2T, D_FF, D_MODEL);

    // ln1 -> h1 (bf16)
    ln_bf16_kernel<<<ROWS, 256, 0, stream>>>(x, ln1w, ln1b, h1);

    // QKV projections (bf16 out)
    gemm_bf16_kernel<0><<<dim3(D_MODEL/64, ROWS/64), 128, 0, stream>>>(
        h1, WqT, bq, nullptr, nullptr, qb_, ROWS, D_MODEL, D_MODEL);
    gemm_bf16_kernel<0><<<dim3(D_MODEL/64, ROWS/64), 128, 0, stream>>>(
        h1, WkT, bk, nullptr, nullptr, kb_, ROWS, D_MODEL, D_MODEL);
    gemm_bf16_kernel<0><<<dim3(D_MODEL/64, ROWS/64), 128, 0, stream>>>(
        h1, WvT, bv, nullptr, nullptr, vb_, ROWS, D_MODEL, D_MODEL);

    // windowed causal attention -> ctx (bf16)
    attn_kernel<<<dim3(SEQ_L/16, N_HEADS, BATCH), 128, 0, stream>>>(qb_, kb_, vb_, ctb);

    // x1 = x + ctx @ Wo + bo   (f32)
    gemm_bf16_kernel<1><<<dim3(D_MODEL/64, ROWS/64), 128, 0, stream>>>(
        ctb, WoT, bo, x, x1, nullptr, ROWS, D_MODEL, D_MODEL);

    // ln2 -> h2 (bf16)
    ln_bf16_kernel<<<ROWS, 256, 0, stream>>>(x1, ln2w, ln2b, h2);

    // g = gelu(h2 @ W1 + b1)   (bf16)
    gemm_bf16_kernel<2><<<dim3(D_FF/64, ROWS/64), 128, 0, stream>>>(
        h2, W1T, b1, nullptr, nullptr, gb_, ROWS, D_FF, D_MODEL);

    // out = x1 + g @ W2 + b2   (f32)
    gemm_bf16_kernel<1><<<dim3(D_MODEL/64, ROWS/64), 128, 0, stream>>>(
        gb_, W2T, b2, x1, (float*)d_out, nullptr, ROWS, D_MODEL, D_FF);
}
